// AttentionManual_6133213299371
// MI455X (gfx1250) — matrix operand order repaired and checked by reading, not compiled, not run on hardware
//
#include <hip/hip_runtime.h>
#include <stdint.h>

#ifndef __has_builtin
#define __has_builtin(x) 0
#endif

// ---- CDNA5 (gfx1250) wave32 WMMA flash attention, LDS-staged K/V ----
// Qt, Kt : (8192 x 128) bf16 row-major (transposed from input (d, s))
// Vc     : (128 x 8192) bf16 (straight convert)
// Per workgroup: async-DMA K-block (64x128) and V-block (128x64) into LDS,
// double-buffered on ASYNCcnt; all WMMA B fragments come from LDS.

typedef __attribute__((ext_vector_type(16))) __bf16 v16bf;
typedef __attribute__((ext_vector_type(8)))  __bf16 v8bf;
typedef __attribute__((ext_vector_type(8)))  float  v8f;
typedef __attribute__((ext_vector_type(4)))  int    v4i;

// address-space-qualified int4 for the async-to-LDS builtin
typedef __attribute__((address_space(1))) v4i gv4i;   // global
typedef __attribute__((address_space(3))) v4i lv4i;   // LDS

#define D_HEAD   128
#define SEQ      8192
#define BN       64      // kv block
#define WAVES    8       // waves per workgroup; each wave owns 16 q rows
#define NTHREADS 256

// LDS layout in halves: K[2][BN*128] | V[2][128*BN] | P[WAVES*16*BN]
#define K_OFF(b)    ((b) * (BN * D_HEAD))
#define V_OFF(b)    (2 * BN * D_HEAD + (b) * (D_HEAD * BN))
#define P_OFF       (4 * BN * D_HEAD)
#define SMEM_HALVES (4 * BN * D_HEAD + WAVES * 16 * BN)   // 40960 halves = 80 KB

static __device__ __forceinline__ unsigned short f32_to_bf16_bits(float f) {
    uint32_t u = __float_as_uint(f);
    uint32_t r = u + 0x7FFFu + ((u >> 16) & 1u);   // round-to-nearest-even
    return (unsigned short)(r >> 16);
}

// 16-byte global -> LDS copy, async (ASYNCcnt) when the gfx1250 builtin exists.
static __device__ __forceinline__ void async_ld16(unsigned short* l, const unsigned short* g) {
#if __has_builtin(__builtin_amdgcn_global_load_async_to_lds_b128)
    __builtin_amdgcn_global_load_async_to_lds_b128(
        (gv4i*)g,        // global src (addrspace 1), const cast away
        (lv4i*)l,        // LDS dst (addrspace 3)
        0, 0);           // imm offset, imm cpol
#else
    *(v8bf*)l = *(const v8bf*)g;     // fallback: synchronous copy through VGPRs
#endif
}

#if __has_builtin(__builtin_amdgcn_s_wait_asynccnt)
#define WAIT_ASYNC(N) __builtin_amdgcn_s_wait_asynccnt(N)
#elif __has_builtin(__builtin_amdgcn_global_load_async_to_lds_b128)
#define WAIT_ASYNC(N) __asm__ volatile("s_wait_asynccnt %0" :: "i"(N) : "memory")
#else
#define WAIT_ASYNC(N) __asm__ volatile("" ::: "memory")
#endif

// Stage one kv block: K rows are contiguous (linear 16 KB); V rows are 128 B slices.
static __device__ __forceinline__ void stage_kv(const unsigned short* __restrict__ Kt,
                                                const unsigned short* __restrict__ Vc,
                                                unsigned short* __restrict__ Kl,
                                                unsigned short* __restrict__ Vl,
                                                int jb, int tid) {
    const unsigned short* kg = Kt + (size_t)jb * D_HEAD;
    #pragma unroll
    for (int i = 0; i < 4; ++i) {
        int c = tid + NTHREADS * i;            // 0..1023 chunks of 8 halves
        async_ld16(Kl + c * 8, kg + c * 8);
    }
    #pragma unroll
    for (int i = 0; i < 4; ++i) {
        int c = tid + NTHREADS * i;            // 0..1023
        int d = c >> 3, cc = c & 7;            // row d (0..127), chunk cc (0..7)
        async_ld16(Vl + d * BN + cc * 8, Vc + (size_t)d * SEQ + jb + cc * 8);
    }
}

__global__ __launch_bounds__(NTHREADS)
void convert_kernel(const float* __restrict__ q,
                    const float* __restrict__ k,
                    const float* __restrict__ v,
                    unsigned short* __restrict__ Qt,
                    unsigned short* __restrict__ Kt,
                    unsigned short* __restrict__ Vc) {
    int i = blockIdx.x * blockDim.x + threadIdx.x;   // < SEQ * D_HEAD
    int s = i >> 7;
    int d = i & 127;
    Qt[i] = f32_to_bf16_bits(q[d * SEQ + s]);
    Kt[i] = f32_to_bf16_bits(k[d * SEQ + s]);
    Vc[i] = f32_to_bf16_bits(v[i]);
}

#define WMMA_BF16(A, B, C) \
    __builtin_amdgcn_wmma_f32_16x16x32_bf16(false, (A), false, (B), (short)0, (C), false, false)

__global__ __launch_bounds__(NTHREADS)
void flash_kernel(const unsigned short* __restrict__ Qt,
                  const unsigned short* __restrict__ Kt,
                  const unsigned short* __restrict__ Vc,
                  float* __restrict__ out) {
    extern __shared__ unsigned short smem[];          // 80 KB dynamic LDS

    const int tid    = threadIdx.x;
    const int lane   = tid & 31;
    const int wave   = tid >> 5;
    const int m0     = (blockIdx.x * WAVES + wave) * 16;
    const int n16    = lane & 15;
    const bool lo    = lane < 16;
    const int rowoff = lo ? 0 : 8;     // f32 C-layout row split across lane halves
    const int kb     = lo ? 0 : 16;    // 16-bit A-layout K base per lane half (contiguous, matches B)
    const int koff   = lo ? 0 : 16;    // 16-bit B-layout K base per lane half

    // --- persistent Q A-fragments: qa[f] covers K(d) = 32f .. 32f+31 ---
    v16bf qa[4];
    {
        const unsigned short* qrow = Qt + (size_t)(m0 + n16) * D_HEAD;
        #pragma unroll
        for (int f = 0; f < 4; ++f) {
            v8bf a = *(const v8bf*)(qrow + 32 * f + kb);
            v8bf b = *(const v8bf*)(qrow + 32 * f + kb + 8);
            #pragma unroll
            for (int i2 = 0; i2 < 8; ++i2) { qa[f][i2] = a[i2]; qa[f][i2 + 8] = b[i2]; }
        }
    }

    v8f zero = {};
    v8f acc[8];
    #pragma unroll
    for (int t = 0; t < 8; ++t) acc[t] = zero;
    float m_i[8], l_i[8];
    #pragma unroll
    for (int r = 0; r < 8; ++r) { m_i[r] = -1e30f; l_i[r] = 0.f; }

    unsigned short* pl = smem + P_OFF + wave * 16 * BN;

    // ---- preload first kv block into buffer 0 ----
    stage_kv(Kt, Vc, smem + K_OFF(0), smem + V_OFF(0), 0, tid);

    for (int ib = 0; ib < SEQ / BN; ++ib) {
        const int jb  = ib * BN;
        const int cur = ib & 1;

        // issue DMA for next block while this one computes
        if (jb + BN < SEQ) {
            stage_kv(Kt, Vc, smem + K_OFF(cur ^ 1), smem + V_OFF(cur ^ 1), jb + BN, tid);
            WAIT_ASYNC(8);     // in-order: current block's 8 chunks have landed
        } else {
            WAIT_ASYNC(0);
        }
        __syncthreads();       // current buffer visible to all waves

        const unsigned short* Klc = smem + K_OFF(cur);
        const unsigned short* Vlc = smem + V_OFF(cur);

        // ---- S = Q * K^T : 16 x BN, f32 accumulate over d=128 ----
        v8f sv[4];
        #pragma unroll
        for (int nt = 0; nt < 4; ++nt) {
            const unsigned short* krow = Klc + (nt * 16 + n16) * D_HEAD + koff;
            v16bf b0 = *(const v16bf*)(krow);
            v16bf b1 = *(const v16bf*)(krow + 32);
            v16bf b2 = *(const v16bf*)(krow + 64);
            v16bf b3 = *(const v16bf*)(krow + 96);
            v8f st = zero;
            st = WMMA_BF16(qa[0], b0, st);
            st = WMMA_BF16(qa[1], b1, st);
            st = WMMA_BF16(qa[2], b2, st);
            st = WMMA_BF16(qa[3], b3, st);
            sv[nt] = st;
        }

        // ---- online softmax (16-lane-half shuffles; wave32) ----
        #pragma unroll
        for (int r = 0; r < 8; ++r) {
            float p0 = sv[0][r], p1 = sv[1][r], p2 = sv[2][r], p3 = sv[3][r];
            float bm = fmaxf(fmaxf(p0, p1), fmaxf(p2, p3));
            #pragma unroll
            for (int off = 8; off >= 1; off >>= 1)
                bm = fmaxf(bm, __shfl_xor(bm, off, 16));
            float mn = fmaxf(m_i[r], bm);
            float sc = __expf(m_i[r] - mn);
            p0 = __expf(p0 - mn); p1 = __expf(p1 - mn);
            p2 = __expf(p2 - mn); p3 = __expf(p3 - mn);
            float bs = (p0 + p1) + (p2 + p3);
            #pragma unroll
            for (int off = 8; off >= 1; off >>= 1)
                bs += __shfl_xor(bs, off, 16);
            l_i[r] = l_i[r] * sc + bs;
            m_i[r] = mn;
            #pragma unroll
            for (int t2 = 0; t2 < 8; ++t2) acc[t2][r] *= sc;
            sv[0][r] = p0; sv[1][r] = p1; sv[2][r] = p2; sv[3][r] = p3;
        }

        // ---- P: f32 C-layout -> bf16 A-layout via wave-private LDS tile ----
        #pragma unroll
        for (int r = 0; r < 8; ++r)
            #pragma unroll
            for (int t = 0; t < 4; ++t)
                pl[(rowoff + r) * BN + t * 16 + n16] = f32_to_bf16_bits(sv[t][r]);
        __asm__ volatile("s_wait_dscnt 0" ::: "memory");   // wave-private; LDS in-order per wave

        v16bf pa[2];
        {
            const unsigned short* prow = pl + n16 * BN;
            #pragma unroll
            for (int kk = 0; kk < 2; ++kk) {
                v8bf a2 = *(const v8bf*)(prow + kk * 32 + kb);
                v8bf b2 = *(const v8bf*)(prow + kk * 32 + kb + 8);
                #pragma unroll
                for (int i2 = 0; i2 < 8; ++i2) { pa[kk][i2] = a2[i2]; pa[kk][i2 + 8] = b2[i2]; }
            }
        }

        // ---- O += P * V : K = 64 kv, N = 128 d ----
        #pragma unroll
        for (int kk = 0; kk < 2; ++kk) {
            #pragma unroll
            for (int dt = 0; dt < 8; dt += 2) {
                v16bf b0 = *(const v16bf*)(Vlc + (dt * 16 + n16) * BN + kk * 32 + koff);
                v16bf b1 = *(const v16bf*)(Vlc + ((dt + 1) * 16 + n16) * BN + kk * 32 + koff);
                acc[dt]     = WMMA_BF16(pa[kk], b0, acc[dt]);
                acc[dt + 1] = WMMA_BF16(pa[kk], b1, acc[dt + 1]);
            }
        }

        __syncthreads();       // everyone done reading cur before it is restaged
    }

    // ---- epilogue ----
    #pragma unroll
    for (int dt = 0; dt < 8; ++dt)
        #pragma unroll
        for (int r = 0; r < 8; ++r)
            out[(size_t)(m0 + rowoff + r) * D_HEAD + dt * 16 + n16] = acc[dt][r] / l_i[r];
}

extern "C" void kernel_launch(void* const* d_in, const int* in_sizes, int n_in,
                              void* d_out, int out_size, void* d_ws, size_t ws_size,
                              hipStream_t stream) {
    (void)in_sizes; (void)n_in; (void)out_size; (void)ws_size;
    const float* q = (const float*)d_in[0];
    const float* k = (const float*)d_in[1];
    const float* v = (const float*)d_in[2];
    float* out = (float*)d_out;

    unsigned short* Qt = (unsigned short*)d_ws;
    unsigned short* Kt = Qt + (size_t)SEQ * D_HEAD;
    unsigned short* Vc = Kt + (size_t)SEQ * D_HEAD;

    const int total = SEQ * D_HEAD;
    convert_kernel<<<total / NTHREADS, NTHREADS, 0, stream>>>(q, k, v, Qt, Kt, Vc);

    const int blocks = SEQ / (16 * WAVES);                 // 64 workgroups
    flash_kernel<<<blocks, NTHREADS, SMEM_HALVES * 2, stream>>>(Qt, Kt, Vc, out);
}
